// Attention_395136991961
// MI455X (gfx1250) — matrix operand order repaired and checked by reading, not compiled, not run on hardware
//
#include <hip/hip_runtime.h>
#include <hip/hip_bf16.h>

// ---------------------------------------------------------------------------
// Types for CDNA5 WMMA (wave32): A/B = 16 bf16 per lane (8 VGPRs),
// C/D = 8 f32 per lane (8 VGPRs).
// ---------------------------------------------------------------------------
typedef __bf16 bf16_t;
typedef __attribute__((ext_vector_type(16))) __bf16 v16bf;
typedef __attribute__((ext_vector_type(8)))  __bf16 v8bf;
typedef __attribute__((ext_vector_type(8)))  float  v8f;

__device__ __forceinline__ bf16_t f32_to_bf16(float f) {
    unsigned u = __builtin_bit_cast(unsigned, f);
    u += 0x7FFFu + ((u >> 16) & 1u);              // round-to-nearest-even
    unsigned short s = (unsigned short)(u >> 16);
    return __builtin_bit_cast(bf16_t, s);
}

// ---------------------------------------------------------------------------
// Fragment loaders per ISA 7.12.2 layouts.
// A (16x32 bf16, M x K): lane L holds row M=L%16.
//   lanes 0-15 : elems 0..7 -> K=0..7,  elems 8..15 -> K=16..23
//   lanes 16-31: elems 0..7 -> K=8..15, elems 8..15 -> K=24..31
// ---------------------------------------------------------------------------
__device__ __forceinline__ v16bf load_a_frag(const bf16_t* __restrict__ base,
                                             int ld, int r0, int k0, int lane) {
    const int m  = lane & 15;
    const int kh = (lane >> 4) << 3;              // 0 or 8
    const bf16_t* p = base + (size_t)(r0 + m) * ld + k0 + kh;
    v8bf lo = *(const v8bf*)p;                    // K = kh + 0..7
    v8bf hi = *(const v8bf*)(p + 16);             // K = kh + 16..23
    v16bf a;
#pragma unroll
    for (int i = 0; i < 8; ++i) { a[i] = lo[i]; a[i + 8] = hi[i]; }
    return a;
}

// B (32x16 bf16, K x N) loaded from ROW-MAJOR W[n][k] (i.e. B = W^T):
// lane L holds column N=L%16; same K layout as A:
// lanes 0-15: elems 0..7 -> K=0..7, 8..15 -> K=16..23; lanes 16-31: K=8..15, K=24..31.
__device__ __forceinline__ v16bf load_bT_frag(const bf16_t* __restrict__ w,
                                              int ld, int n0, int k0, int lane) {
    const int n  = lane & 15;
    const int kh = (lane >> 4) << 3;              // 0 or 8
    const bf16_t* p = w + (size_t)(n0 + n) * ld + k0 + kh;
    v8bf c0 = *(const v8bf*)p;
    v8bf c1 = *(const v8bf*)(p + 16);
    v16bf b;
#pragma unroll
    for (int i = 0; i < 8; ++i) { b[i] = c0[i]; b[i + 8] = c1[i]; }
    return b;
}

#define WMMA_BF16(A, B, C) \
    __builtin_amdgcn_wmma_f32_16x16x32_bf16(false, (A), false, (B), (short)0, (C), false, false)

// ---------------------------------------------------------------------------
// Kernel 0: fp32 -> bf16 convert (prepass for x, qkv_w, proj_w)
// ---------------------------------------------------------------------------
__global__ void cvt_f32_bf16_kernel(const float* __restrict__ in,
                                    bf16_t* __restrict__ out, int n) {
    int i = blockIdx.x * blockDim.x + threadIdx.x;
    if (i < n) out[i] = f32_to_bf16(in[i]);
}

// ---------------------------------------------------------------------------
// Kernel 1: QKV GEMM (8192x3072x1024) + fused RoPE + 1/sqrt(D) scale.
// Grid: (48 col blocks, 128 row blocks), 128 threads (4 waves).
// Each wave: 16 rows x 64 cols (4 accumulator tiles). Col block == one
// (which,head) slice, so RoPE partner d<->d+32 is acc[j] <-> acc[j^2].
// Q,K stored [bh][tok][64]; V stored TRANSPOSED [bh][d][1024] so the
// attention P*V B-fragments become contiguous loads.
// ---------------------------------------------------------------------------
__global__ __launch_bounds__(128) void qkv_rope_kernel(
    const bf16_t* __restrict__ xb,     // [8192][1024]
    const bf16_t* __restrict__ wb,     // [3072][1024]
    const float*  __restrict__ rope,   // [1024][64]
    bf16_t* __restrict__ qb,           // [(b*16+h)][1024][64]
    bf16_t* __restrict__ kb,           // [(b*16+h)][1024][64]
    bf16_t* __restrict__ vt) {         // [(b*16+h)][64][1024]  (transposed!)
    const int lane  = threadIdx.x & 31;
    const int wave  = threadIdx.x >> 5;
    const int cb    = blockIdx.x;                 // 0..47
    const int row0  = blockIdx.y * 64 + wave * 16;
    const int col0  = cb * 64;
    const int which = cb >> 4;                    // 0=q 1=k 2=v
    const int head  = cb & 15;

    v8f acc[4] = {{}, {}, {}, {}};
    for (int k0 = 0; k0 < 1024; k0 += 32) {
        v16bf a = load_a_frag(xb, 1024, row0, k0, lane);
#pragma unroll
        for (int j = 0; j < 4; ++j) {
            v16bf b = load_bT_frag(wb, 1024, col0 + j * 16, k0, lane);
            acc[j] = WMMA_BF16(a, b, acc[j]);
        }
    }

    const int nlo = lane & 15;                    // N within tile (feature d%16)
    const int hi  = lane >> 4;                    // row half
    if (which == 2) {
        // ---- V: transposed store; per lane rows r map to consecutive tokens,
        // so each tile collapses to one 16-byte global_store_b128.
        const int row_base = row0 + 8 * hi;       // 8 consecutive tokens
        const int b   = row_base >> 10;
        const int tok = row_base & 1023;
#pragma unroll
        for (int j = 0; j < 4; ++j) {
            const int d = j * 16 + nlo;
            v8bf pack;
#pragma unroll
            for (int r = 0; r < 8; ++r) pack[r] = f32_to_bf16(acc[j][r]);
            *(v8bf*)(vt + ((size_t)(b * 16 + head) * 64 + d) * 1024 + tok) = pack;
        }
    } else {                                      // ---- Q/K: fused RoPE
        bf16_t* dst = (which == 0) ? qb : kb;
        const float sc = (which == 0) ? 0.125f : 1.0f;   // fold scale=D^-0.5 into Q
#pragma unroll
        for (int j = 0; j < 4; ++j)
#pragma unroll
            for (int r = 0; r < 8; ++r) {
                int row = row0 + r + 8 * hi;
                int b = row >> 10, tok = row & 1023;
                int d = j * 16 + nlo;
                float v  = acc[j][r];
                float vp = acc[j ^ 2][r];          // partner element d +/- 32
                float rot = (j < 2) ? -vp : vp;    // rotate_half
                float s, c;
                __sincosf(rope[tok * 64 + d], &s, &c);
                dst[((size_t)(b * 16 + head) * 1024 + tok) * 64 + d] =
                    f32_to_bf16((v * c + rot * s) * sc);
            }
    }
}

// ---------------------------------------------------------------------------
// Kernel 2: flash attention per (b,h). Grid: (128 bh, 8 qblocks), 128 thr.
// Wave = 32 query rows (2 row tiles). Keys streamed in blocks of 64 with
// online softmax; P re-shaped via wave-private LDS tile for the P*V WMMA.
// V is pre-transposed [d][tok] so its B-fragments are contiguous loads.
// ---------------------------------------------------------------------------
__global__ __launch_bounds__(128) void attn_kernel(
    const bf16_t* __restrict__ qb, const bf16_t* __restrict__ kb,
    const bf16_t* __restrict__ vt,
    bf16_t* __restrict__ ob) {                    // [8][1024][1024] (b,n,h*64+d)
    __shared__ __align__(16) bf16_t pshare[4][32][72];   // per-wave P tile, padded

    const int lane = threadIdx.x & 31;
    const int wave = threadIdx.x >> 5;
    const int bh   = blockIdx.x;
    const int b = bh >> 4, h = bh & 15;
    const bf16_t* qh  = qb + (size_t)bh * 1024 * 64;
    const bf16_t* kh  = kb + (size_t)bh * 1024 * 64;
    const bf16_t* vth = vt + (size_t)bh * 64 * 1024;
    const int qrow0 = blockIdx.y * 128 + wave * 32;
    const int nlo = lane & 15, hi = lane >> 4;
    bf16_t* pw = &pshare[wave][0][0];

    v16bf qa[2][2];
#pragma unroll
    for (int t = 0; t < 2; ++t)
#pragma unroll
        for (int kk = 0; kk < 2; ++kk)
            qa[t][kk] = load_a_frag(qh, 64, qrow0 + t * 16, kk * 32, lane);

    v8f o[2][4] = {{{}, {}, {}, {}}, {{}, {}, {}, {}}};
    float mrow[2][8], lrow[2][8];
#pragma unroll
    for (int t = 0; t < 2; ++t)
#pragma unroll
        for (int r = 0; r < 8; ++r) { mrow[t][r] = -3.0e38f; lrow[t][r] = 0.0f; }

    for (int key0 = 0; key0 < 1024; key0 += 64) {
        // Prefetch next key block's K rows / V^T columns into cache
        // (global_prefetch_b8; speculative, tail over-reach is harmless).
        {
            const int nxt = key0 + 64;
            __builtin_prefetch(kh + (size_t)(nxt + (lane & 63)) * 64, 0, 1);
            __builtin_prefetch(vth + (size_t)(lane & 63) * 1024 + nxt, 0, 1);
        }
        // ---- S = Q * K^T  (32 x 64 logits)
        v8f s[2][4] = {{{}, {}, {}, {}}, {{}, {}, {}, {}}};
#pragma unroll
        for (int kk = 0; kk < 2; ++kk) {
            v16bf bkf[4];
#pragma unroll
            for (int j = 0; j < 4; ++j)
                bkf[j] = load_bT_frag(kh, 64, key0 + j * 16, kk * 32, lane);
#pragma unroll
            for (int t = 0; t < 2; ++t)
#pragma unroll
                for (int j = 0; j < 4; ++j)
                    s[t][j] = WMMA_BF16(qa[t][kk], bkf[j], s[t][j]);
        }
        // ---- online softmax update (rows span the 16-lane halves)
#pragma unroll
        for (int t = 0; t < 2; ++t)
#pragma unroll
            for (int r = 0; r < 8; ++r) {
                float mx = fmaxf(fmaxf(s[t][0][r], s[t][1][r]),
                                 fmaxf(s[t][2][r], s[t][3][r]));
                mx = fmaxf(mx, __shfl_xor(mx, 1, 32));
                mx = fmaxf(mx, __shfl_xor(mx, 2, 32));
                mx = fmaxf(mx, __shfl_xor(mx, 4, 32));
                mx = fmaxf(mx, __shfl_xor(mx, 8, 32));
                float mnew = fmaxf(mrow[t][r], mx);
                float corr = __expf(mrow[t][r] - mnew);
                mrow[t][r] = mnew;
                float psum = 0.0f;
#pragma unroll
                for (int j = 0; j < 4; ++j) {
                    float p = __expf(s[t][j][r] - mnew);
                    s[t][j][r] = p;
                    psum += p;
                }
                psum += __shfl_xor(psum, 1, 32);
                psum += __shfl_xor(psum, 2, 32);
                psum += __shfl_xor(psum, 4, 32);
                psum += __shfl_xor(psum, 8, 32);
                lrow[t][r] = lrow[t][r] * corr + psum;
#pragma unroll
                for (int j = 0; j < 4; ++j) o[t][j][r] *= corr;
                const int mr = t * 16 + r + 8 * hi;
#pragma unroll
                for (int j = 0; j < 4; ++j)
                    pw[mr * 72 + j * 16 + nlo] = f32_to_bf16(s[t][j][r]);
            }
        asm volatile("s_wait_dscnt 0x0" ::: "memory");   // P stores visible (LDS in-order per wave)
        // ---- O += P * V   (B[k=key][n=d] = vt[d][key] -> contiguous loads)
#pragma unroll
        for (int kk = 0; kk < 2; ++kk) {
            v16bf bvf[4];
#pragma unroll
            for (int j = 0; j < 4; ++j)
                bvf[j] = load_bT_frag(vth, 1024, j * 16, key0 + kk * 32, lane);
#pragma unroll
            for (int t = 0; t < 2; ++t) {
                v16bf pa = load_a_frag(pw, 72, t * 16, kk * 32, lane);
#pragma unroll
                for (int j = 0; j < 4; ++j)
                    o[t][j] = WMMA_BF16(pa, bvf[j], o[t][j]);
            }
        }
    }
    // ---- normalize + store to (b, n, h*64+d) bf16
#pragma unroll
    for (int t = 0; t < 2; ++t)
#pragma unroll
        for (int j = 0; j < 4; ++j)
#pragma unroll
            for (int r = 0; r < 8; ++r) {
                int tok = qrow0 + t * 16 + r + 8 * hi;
                int d   = j * 16 + nlo;
                ob[((size_t)b * 1024 + tok) * 1024 + h * 64 + d] =
                    f32_to_bf16(o[t][j][r] / lrow[t][r]);
            }
}

// ---------------------------------------------------------------------------
// Kernel 3: output projection GEMM (8192x1024x1024) + bias, fp32 out.
// Grid: (16 col blocks, 128 row blocks), 128 threads.
// ---------------------------------------------------------------------------
__global__ __launch_bounds__(128) void proj_kernel(
    const bf16_t* __restrict__ ab,     // [8192][1024]
    const bf16_t* __restrict__ wb,     // [1024][1024]
    const float*  __restrict__ bias,   // [1024]
    float* __restrict__ out) {         // [8192][1024]
    const int lane = threadIdx.x & 31;
    const int wave = threadIdx.x >> 5;
    const int row0 = blockIdx.y * 64 + wave * 16;
    const int col0 = blockIdx.x * 64;

    v8f acc[4] = {{}, {}, {}, {}};
    for (int k0 = 0; k0 < 1024; k0 += 32) {
        v16bf a = load_a_frag(ab, 1024, row0, k0, lane);
#pragma unroll
        for (int j = 0; j < 4; ++j) {
            v16bf b = load_bT_frag(wb, 1024, col0 + j * 16, k0, lane);
            acc[j] = WMMA_BF16(a, b, acc[j]);
        }
    }
    const int nlo = lane & 15, hi = lane >> 4;
#pragma unroll
    for (int j = 0; j < 4; ++j) {
        const int col = col0 + j * 16 + nlo;
        const float bv = bias[col];
#pragma unroll
        for (int r = 0; r < 8; ++r) {
            int row = row0 + r + 8 * hi;
            out[(size_t)row * 1024 + col] = acc[j][r] + bv;
        }
    }
}

// ---------------------------------------------------------------------------
extern "C" void kernel_launch(void* const* d_in, const int* in_sizes, int n_in,
                              void* d_out, int out_size, void* d_ws, size_t ws_size,
                              hipStream_t stream) {
    const float* x      = (const float*)d_in[0];   // [8,1024,1024]
    const float* rope   = (const float*)d_in[1];   // [1,1024,1,64]
    const float* qkv_w  = (const float*)d_in[2];   // [3072,1024]
    const float* proj_w = (const float*)d_in[3];   // [1024,1024]
    const float* proj_b = (const float*)d_in[4];   // [1024]
    float* out = (float*)d_out;

    char* ws = (char*)d_ws;
    size_t off = 0;
    bf16_t* xb    = (bf16_t*)(ws + off); off += (size_t)8192 * 1024 * 2;
    bf16_t* wqkv  = (bf16_t*)(ws + off); off += (size_t)3072 * 1024 * 2;
    bf16_t* wproj = (bf16_t*)(ws + off); off += (size_t)1024 * 1024 * 2;
    bf16_t* qbuf  = (bf16_t*)(ws + off); off += (size_t)128 * 1024 * 64 * 2;
    bf16_t* kbuf  = (bf16_t*)(ws + off); off += (size_t)128 * 1024 * 64 * 2;
    bf16_t* vtbuf = (bf16_t*)(ws + off); off += (size_t)128 * 64 * 1024 * 2;
    bf16_t* abuf  = (bf16_t*)(ws + off); off += (size_t)8192 * 1024 * 2;

    const int nx = 8192 * 1024, nw = 3072 * 1024, np = 1024 * 1024;
    cvt_f32_bf16_kernel<<<(nx + 255) / 256, 256, 0, stream>>>(x, xb, nx);
    cvt_f32_bf16_kernel<<<(nw + 255) / 256, 256, 0, stream>>>(qkv_w, wqkv, nw);
    cvt_f32_bf16_kernel<<<(np + 255) / 256, 256, 0, stream>>>(proj_w, wproj, np);

    qkv_rope_kernel<<<dim3(48, 128), 128, 0, stream>>>(xb, wqkv, rope, qbuf, kbuf, vtbuf);
    attn_kernel<<<dim3(128, 8), 128, 0, stream>>>(qbuf, kbuf, vtbuf, abuf);
    proj_kernel<<<dim3(16, 128), 128, 0, stream>>>(abuf, wproj, proj_b, out);
}
